// LSTMDecoder_3238405341196
// MI455X (gfx1250) — matrix operand order repaired and checked by reading, not compiled, not run on hardware
//
#include <hip/hip_runtime.h>
#include <math.h>

// ---------------------------------------------------------------- types
typedef __bf16 bf16_t;
typedef __attribute__((ext_vector_type(16))) __bf16 v16bf;
typedef __attribute__((ext_vector_type(8)))  float  v8f;

#define B_   64
#define P_   196
#define E_   2048
#define D_   512
#define A_   512
#define EMB_ 512
#define V_   10000
#define T_   40
#define XDIM 2560   // EMB + E
#define G4D  2048   // 4*D

// ---------------------------------------------------------------- GEMM (bf16 WMMA, f32 accum)
// C[M,N] (+)= A_bf16[M x K] @ Bt_bf16[N x K]^T (+ bias).  K % 32 == 0, Mtiles % 4 == 0.
// One wave computes a 64-row x 16-col stripe (4 m-tiles): the B fragment is loaded once
// per k-step and reused by 4 independent WMMAs (hides XDL latency, 4x less B traffic).
// Both fragments are single contiguous 32B loads (2x global_load_b128), no converts.
//
// A-layout (ISA 16-bit A 16x32): lane<16: M=lane, K slots {h..h+7, h+16..h+23}, h=8*hi.
//   This code uses k = 16*hi + s instead (as B): with A row-major bf16, two 16B runs at +16*hi, +16*hi+8:
//   element order v[0..7] = k 16hi..16hi+7, v[8..15] = k 16hi+8..16hi+15 -> one v16bf whose
//   two halves we load from (row, 16*hi) and (row, 16*hi+8).
// B-layout (dense 32x16, same contiguous k order as A): lane<16: N=lane, K=0..15; lane>=16: K=16..31:
//   one contiguous 32B run at (col, 16*hi).
__device__ __forceinline__ v16bf load_a_frag(const bf16_t* ap)
{
    typedef __attribute__((ext_vector_type(8))) __bf16 v8bf;
    v8bf lo = *(const v8bf*)(ap);
    v8bf hi = *(const v8bf*)(ap + 8);
    v16bf r;
#pragma unroll
    for (int j = 0; j < 8; ++j) { r[j] = lo[j]; r[8 + j] = hi[j]; }
    return r;
}

__global__ void gemm_bf16_wmma(const bf16_t* __restrict__ A, int lda,
                               const bf16_t* __restrict__ Bt, int ldb,
                               const float* __restrict__ bias,
                               float* __restrict__ C, int ldc,
                               int Mgroups, int Ntiles, int K,
                               int accum, int store_bf16)
{
    int wid = blockIdx.x * (blockDim.x >> 5) + (threadIdx.x >> 5);
    if (wid >= Mgroups * Ntiles) return;          // wave-uniform: EXEC stays all-ones
    int mg = wid / Ntiles;
    int nt = wid % Ntiles;
    int lane = threadIdx.x & 31;
    int l15  = lane & 15;
    int hi   = lane >> 4;                         // 0 or 1

    const bf16_t* Abase = A  + (size_t)(mg * 64 + l15) * lda + hi * 16;
    const bf16_t* Brow  = Bt + (size_t)(nt * 16 + l15) * ldb + hi * 16;

    v8f acc0 = {}, acc1 = {}, acc2 = {}, acc3 = {};
    size_t mstride = (size_t)16 * lda;
#pragma unroll 2
    for (int k0 = 0; k0 < K; k0 += 32) {
        v16bf bv = *(const v16bf*)(Brow + k0);    // 32B contiguous
        const bf16_t* ap = Abase + k0;
        v16bf a0 = load_a_frag(ap);
        v16bf a1 = load_a_frag(ap + mstride);
        v16bf a2 = load_a_frag(ap + 2 * mstride);
        v16bf a3 = load_a_frag(ap + 3 * mstride);
        acc0 = __builtin_amdgcn_wmma_f32_16x16x32_bf16(false, a0, false, bv, (short)0, acc0, false, false);
        acc1 = __builtin_amdgcn_wmma_f32_16x16x32_bf16(false, a1, false, bv, (short)0, acc1, false, false);
        acc2 = __builtin_amdgcn_wmma_f32_16x16x32_bf16(false, a2, false, bv, (short)0, acc2, false, false);
        acc3 = __builtin_amdgcn_wmma_f32_16x16x32_bf16(false, a3, false, bv, (short)0, acc3, false, false);
    }

    // C/D layout: lane -> col = nt*16 + l15; rows = mtile*16 + 8*hi + r
    int col = nt * 16 + l15;
    float bval = bias ? bias[col] : 0.0f;
    v8f accs[4] = {acc0, acc1, acc2, acc3};
#pragma unroll
    for (int mb = 0; mb < 4; ++mb) {
        int row0 = mg * 64 + mb * 16 + hi * 8;
#pragma unroll
        for (int r = 0; r < 8; ++r) {
            size_t off = (size_t)(row0 + r) * ldc + col;
            float v = accs[mb][r] + bval;
            if (store_bf16) {
                ((bf16_t*)C)[off] = (bf16_t)v;
            } else {
                if (accum) v += C[off];
                C[off] = v;
            }
        }
    }
}

// ---------------------------------------------------------------- weight transpose+convert
// W_f32[K x N] row-major -> Wt_bf16[N x K] row-major
__global__ void conv_t_kernel(const float* __restrict__ W, bf16_t* __restrict__ Wt,
                              int K, int N)
{
    size_t idx = (size_t)blockIdx.x * 256 + threadIdx.x;
    if (idx >= (size_t)K * N) return;
    size_t n = idx / K, k = idx % K;
    Wt[idx] = (bf16_t)W[k * (size_t)N + n];
}

// ---------------------------------------------------------------- plain f32 -> bf16 copy
__global__ void conv_kernel(const float* __restrict__ W, bf16_t* __restrict__ Wb, size_t n)
{
    size_t idx = (size_t)blockIdx.x * 256 + threadIdx.x;
    if (idx < n) Wb[idx] = (bf16_t)W[idx];
}

// ---------------------------------------------------------------- bias prep
__global__ void prep_bias(const float* __restrict__ dab, const float* __restrict__ fbb,
                          const float* __restrict__ bih, const float* __restrict__ bhh,
                          float* __restrict__ bcat, float* __restrict__ bgate)
{
    int i = blockIdx.x * 256 + threadIdx.x;
    if (i < A_)                bcat[i] = dab[i];
    if (i >= A_ && i < XDIM)   bcat[i] = fbb[i - A_];
    if (i < G4D)               bgate[i] = bih[i] + bhh[i];
}

// ---------------------------------------------------------------- mean pool over P (-> bf16)
__global__ void mean_pool(const float* __restrict__ enc, bf16_t* __restrict__ mean_enc)
{
    int b = blockIdx.x, tid = threadIdx.x;
    const float* e = enc + (size_t)b * P_ * E_;
    for (int j = tid; j < E_; j += 256) {
        float s = 0.f;
        for (int p = 0; p < P_; ++p) s += e[(size_t)p * E_ + j];
        mean_enc[(size_t)b * E_ + j] = (bf16_t)(s * (1.0f / (float)P_));
    }
}

// ---------------------------------------------------------------- per-sample attention step
// One workgroup per batch element: scores -> softmax -> context -> x = [emb | gate*ctx]
__global__ void attn_step(const float* __restrict__ att1,
                          const float* __restrict__ hcat,      // [64, 2560]: att2 | gate_pre
                          const float* __restrict__ enc,
                          const float* __restrict__ embedding,
                          const int*   __restrict__ captions,
                          const float* __restrict__ full_att_W,
                          const float* __restrict__ full_att_b,
                          float* __restrict__ alphas,          // + t*P_; row stride T_*P_
                          bf16_t* __restrict__ x,              // [64, 2560] bf16
                          int t)
{
    __shared__ float att2s[A_];
    __shared__ float es[P_];
    __shared__ float red[16];
    int b = blockIdx.x, tid = threadIdx.x;
    int lane = tid & 31, wave = tid >> 5;

    for (int i = tid; i < A_; i += 256) att2s[i] = hcat[(size_t)b * XDIM + i];
    __syncthreads();

    // e[p] = relu(att1[b,p,:] + att2) . full_att_W  (one wave per p)
    for (int p = wave; p < P_; p += 8) {
        const float* a1 = att1 + ((size_t)b * P_ + p) * A_;
        float acc = 0.f;
        for (int k = lane; k < A_; k += 32) {
            float v = a1[k] + att2s[k];
            acc += fmaxf(v, 0.f) * full_att_W[k];
        }
#pragma unroll
        for (int o = 16; o > 0; o >>= 1) acc += __shfl_xor(acc, o);
        if (lane == 0) es[p] = acc + full_att_b[0];
    }
    __syncthreads();

    // softmax over P_
    float lm = -3.4e38f;
    for (int p = tid; p < P_; p += 256) lm = fmaxf(lm, es[p]);
#pragma unroll
    for (int o = 16; o > 0; o >>= 1) lm = fmaxf(lm, __shfl_xor(lm, o));
    if (lane == 0) red[wave] = lm;
    __syncthreads();
    if (tid == 0) {
        float m = red[0];
        for (int i = 1; i < 8; ++i) m = fmaxf(m, red[i]);
        red[8] = m;
    }
    __syncthreads();
    float M = red[8];
    float ls = 0.f;
    for (int p = tid; p < P_; p += 256) { float v = __expf(es[p] - M); es[p] = v; ls += v; }
#pragma unroll
    for (int o = 16; o > 0; o >>= 1) ls += __shfl_xor(ls, o);
    if (lane == 0) red[wave] = ls;
    __syncthreads();
    if (tid == 0) {
        float s = 0.f;
        for (int i = 0; i < 8; ++i) s += red[i];
        red[9] = 1.0f / s;
    }
    __syncthreads();
    float inv = red[9];
    for (int p = tid; p < P_; p += 256) {
        float a = es[p] * inv;
        es[p] = a;
        alphas[(size_t)b * (T_ * P_) + p] = a;
    }
    __syncthreads();

    // context[e] = sum_p alpha[p]*enc[b,p,e];  x[:,512+e] = sigmoid(gate_pre)*context (bf16)
    const float* encb = enc + (size_t)b * P_ * E_;
    for (int e = tid; e < E_; e += 256) {
        float acc = 0.f;
#pragma unroll 4
        for (int p = 0; p < P_; ++p) acc += es[p] * encb[(size_t)p * E_ + e];
        float g = hcat[(size_t)b * XDIM + A_ + e];
        g = 1.0f / (1.0f + __expf(-g));
        x[(size_t)b * XDIM + EMB_ + e] = (bf16_t)(g * acc);
    }
    int cap = captions[b * T_ + t];
    for (int j = tid; j < EMB_; j += 256)
        x[(size_t)b * XDIM + j] = (bf16_t)embedding[(size_t)cap * EMB_ + j];
}

// ---------------------------------------------------------------- LSTM cell (pointwise)
// Writes h directly as bf16 (it is only consumed as a GEMM A-operand).
__global__ void lstm_cell(const float* __restrict__ gates, bf16_t* __restrict__ h,
                          float* __restrict__ c)
{
    int idx = blockIdx.x * 256 + threadIdx.x;
    if (idx >= B_ * D_) return;
    int b = idx >> 9, d = idx & (D_ - 1);
    const float* g = gates + (size_t)b * G4D;
    float ig = 1.0f / (1.0f + __expf(-g[d]));
    float fg = 1.0f / (1.0f + __expf(-g[D_ + d]));
    float gg = tanhf(g[2 * D_ + d]);
    float og = 1.0f / (1.0f + __expf(-g[3 * D_ + d]));
    float cn = fg * c[idx] + ig * gg;
    float hn = og * tanhf(cn);
    c[idx] = cn;
    h[idx] = (bf16_t)hn;
}

// ---------------------------------------------------------------- host
static void launch_gemm(const bf16_t* A, int lda, const bf16_t* Bt, int ldb,
                        const float* bias, void* C, int ldc,
                        int Mtiles, int Ntiles, int K, int accum, int store_bf16,
                        hipStream_t s)
{
    int Mgroups = Mtiles / 4;                     // all our GEMMs have Mtiles % 4 == 0
    int waves   = Mgroups * Ntiles;
    int blocks  = (waves + 7) / 8;                // 8 waves (256 threads) per block
    gemm_bf16_wmma<<<blocks, 256, 0, s>>>(A, lda, Bt, ldb, bias, (float*)C, ldc,
                                          Mgroups, Ntiles, K, accum, store_bf16);
}

extern "C" void kernel_launch(void* const* d_in, const int* in_sizes, int n_in,
                              void* d_out, int out_size, void* d_ws, size_t ws_size,
                              hipStream_t stream)
{
    const float* enc        = (const float*)d_in[0];
    const int*   captions   = (const int*)  d_in[1];
    const float* enc_att_W  = (const float*)d_in[3];
    const float* enc_att_b  = (const float*)d_in[4];
    const float* dec_att_W  = (const float*)d_in[5];
    const float* dec_att_b  = (const float*)d_in[6];
    const float* full_att_W = (const float*)d_in[7];
    const float* full_att_b = (const float*)d_in[8];
    const float* embedding  = (const float*)d_in[9];
    const float* W_ih       = (const float*)d_in[10];
    const float* b_ih       = (const float*)d_in[11];
    const float* W_hh       = (const float*)d_in[12];
    const float* b_hh       = (const float*)d_in[13];
    const float* init_h_W   = (const float*)d_in[14];
    const float* init_h_b   = (const float*)d_in[15];
    const float* init_c_W   = (const float*)d_in[16];
    const float* init_c_b   = (const float*)d_in[17];
    const float* fc_W       = (const float*)d_in[18];
    const float* fc_b       = (const float*)d_in[19];
    const float* f_beta_W   = (const float*)d_in[20];
    const float* f_beta_b   = (const float*)d_in[21];

    float* preds  = (float*)d_out;                          // [B,T,V]
    float* alphas = preds + (size_t)B_ * T_ * V_;           // [B,T,P]

    // workspace carve-up (256B aligned)
    char*  ws  = (char*)d_ws;
    size_t off = 0;
    auto alloc = [&](size_t bytes) -> char* {
        char* p = ws + off;
        off = (off + bytes + 255) & ~(size_t)255;
        return p;
    };
    bf16_t* encatt_t = (bf16_t*)alloc((size_t)A_   * E_   * 2); // [512][2048]
    bf16_t* inith_t  = (bf16_t*)alloc((size_t)D_   * E_   * 2);
    bf16_t* initc_t  = (bf16_t*)alloc((size_t)D_   * E_   * 2);
    bf16_t* wcat_t   = (bf16_t*)alloc((size_t)XDIM * D_   * 2); // [2560][512]: dec_att | f_beta
    bf16_t* wih_t    = (bf16_t*)alloc((size_t)G4D  * XDIM * 2); // [2048][2560]
    bf16_t* whh_t    = (bf16_t*)alloc((size_t)G4D  * D_   * 2); // [2048][512]
    bf16_t* fc_t     = (bf16_t*)alloc((size_t)V_   * D_   * 2); // [10000][512]
    bf16_t* enc_b16  = (bf16_t*)alloc((size_t)B_ * P_ * E_ * 2);// encoder_out as bf16
    float*  bcat     = (float*) alloc((size_t)XDIM * 4);
    float*  bgate    = (float*) alloc((size_t)G4D  * 4);
    bf16_t* mean_b   = (bf16_t*)alloc((size_t)B_ * E_   * 2);
    bf16_t* h_b      = (bf16_t*)alloc((size_t)B_ * D_   * 2);
    float*  c        = (float*) alloc((size_t)B_ * D_   * 4);
    float*  hcat     = (float*) alloc((size_t)B_ * XDIM * 4);
    bf16_t* x_b      = (bf16_t*)alloc((size_t)B_ * XDIM * 2);
    float*  gates    = (float*) alloc((size_t)B_ * G4D  * 4);
    float*  att1     = (float*) alloc((size_t)B_ * P_ * A_ * 4);
    (void)in_sizes; (void)n_in; (void)out_size; (void)ws_size;

    auto convT = [&](const float* W, bf16_t* Wt, int K, int N) {
        size_t total = (size_t)K * N;
        conv_t_kernel<<<(int)((total + 255) / 256), 256, 0, stream>>>(W, Wt, K, N);
    };

    // ---- one-time prep (cheap; graph-replayed every call, inputs never mutated)
    convT(enc_att_W, encatt_t, E_, A_);
    convT(init_h_W,  inith_t,  E_, D_);
    convT(init_c_W,  initc_t,  E_, D_);
    convT(dec_att_W, wcat_t,                   D_, A_); // rows 0..511
    convT(f_beta_W,  wcat_t + (size_t)A_ * D_, D_, E_); // rows 512..2559
    convT(W_ih,      wih_t,   XDIM, G4D);
    convT(W_hh,      whh_t,   D_,   G4D);
    convT(fc_W,      fc_t,    D_,   V_);
    {
        size_t n = (size_t)B_ * P_ * E_;
        conv_kernel<<<(int)((n + 255) / 256), 256, 0, stream>>>(enc, enc_b16, n);
    }
    prep_bias<<<(XDIM + 255) / 256, 256, 0, stream>>>(dec_att_b, f_beta_b, b_ih, b_hh,
                                                      bcat, bgate);
    mean_pool<<<B_, 256, 0, stream>>>(enc, mean_b);

    // h0 (bf16) / c0 (f32) = mean_enc @ init_{h,c}_W + b    (M=64, N=512, K=2048)
    launch_gemm(mean_b, E_, inith_t, E_, init_h_b, h_b, D_, B_ / 16, D_ / 16, E_, 0, 1, stream);
    launch_gemm(mean_b, E_, initc_t, E_, init_c_b, c,   D_, B_ / 16, D_ / 16, E_, 0, 0, stream);

    // att1 = encoder_out @ enc_att_W + b       (M=12544, N=512, K=2048) — the big GEMM
    launch_gemm(enc_b16, E_, encatt_t, E_, enc_att_b, att1, A_,
                (B_ * P_) / 16, A_ / 16, E_, 0, 0, stream);

    // ---- 40 sequential decode steps
    for (int t = 0; t < T_; ++t) {
        // hcat = h @ [dec_att_W | f_beta_W] + [b]   (M=64, N=2560, K=512)
        launch_gemm(h_b, D_, wcat_t, D_, bcat, hcat, XDIM, B_ / 16, XDIM / 16, D_, 0, 0, stream);

        // attention + softmax + context + build x (one WG per batch element)
        attn_step<<<B_, 256, 0, stream>>>(att1, hcat, enc, embedding, captions,
                                          full_att_W, full_att_b,
                                          alphas + (size_t)t * P_, x_b, t);

        // gates = x @ W_ih + (b_ih+b_hh)           (M=64, N=2048, K=2560)
        launch_gemm(x_b, XDIM, wih_t, XDIM, bgate, gates, G4D, B_ / 16, G4D / 16, XDIM, 0, 0, stream);
        // gates += h @ W_hh                        (M=64, N=2048, K=512)
        launch_gemm(h_b, D_, whh_t, D_, nullptr, gates, G4D, B_ / 16, G4D / 16, D_, 1, 0, stream);

        // LSTM pointwise: updates c (f32) and h (bf16) in place
        lstm_cell<<<(B_ * D_ + 255) / 256, 256, 0, stream>>>(gates, h_b, c);

        // preds[:, t, :] = h_new @ fc_W + fc_b     (M=64, N=10000, K=512)
        launch_gemm(h_b, D_, fc_t, D_, fc_b, preds + (size_t)t * V_, T_ * V_,
                    B_ / 16, V_ / 16, D_, 0, 0, stream);
    }
}
